// Interaction_46102178956022
// MI455X (gfx1250) — matrix operand order repaired and checked by reading, not compiled, not run on hardware
//
#include <hip/hip_runtime.h>
#include <hip/hip_bf16.h>

// Box-attention on MI455X (gfx1250, wave32).
// q/k/v: [32 boxes][4 batch][64 ch][80][80] f32; out same shape as v.
// Per pixel: S = Q K^T / 8 (32x32 over 64 ch), softmax over source box j, O = W V.
// HBM-bound (~840 MB @ 23.3 TB/s): coalesced b128 global traffic; f16 LDS slabs
// laid out so every WMMA fragment is two contiguous ds_load_b128.
// Both GEMMs are computed TRANSPOSED (S^T = K Q^T, O^T = V^T W^T) so the
// softmax axis j lives in VGPR rows: the max/sum reductions are register ops
// plus one immediate-pattern ds_swizzle (xor 16) each -- no bpermute chains.

typedef __attribute__((ext_vector_type(16))) _Float16 v16h;
typedef __attribute__((ext_vector_type(8)))  _Float16 v8h;
typedef __attribute__((ext_vector_type(2)))  _Float16 h2;
typedef __attribute__((ext_vector_type(8)))  float    v8f;
typedef __attribute__((ext_vector_type(4)))  float    v4f;

#define NBOX   32
#define NBATCH 4
#define CH     64
#define HW     6400          // 80*80
#define PIX    4             // pixels per block: 16B contiguous per (box,ch) row
#define NTHREADS 128         // 4 waves; one wave per pixel

// LDS layout:
//  Qh [32][4][64] f16 : 16KB  (ch contiguous)
//  Kh [32][4][64] f16 : 16KB  (ch contiguous)
//  Vt [64][4][32] f16 : 16KB  (box contiguous)
//  Wb [4][32][32] f16 :  8KB  (E^T, j contiguous)
//  Os [2048][4]   f32 : 32KB  (output staging for coalesced b128 stores)
// total 88KB -> 3 blocks (12 waves) per WGP.
#define QH_HALF (NBOX * PIX * CH)                  // 8192 halves
#define OS_OFF  (3 * QH_HALF + PIX * NBOX * NBOX)  // in halves

// lane' = lane ^ m via ds_swizzle group-of-32 (and=0x1f, or=0, xor=m), imm only.
#define SWIZ_XOR(x, m) \
    __int_as_float(__builtin_amdgcn_ds_swizzle(__float_as_int(x), ((m) << 10) | 0x1f))

__device__ inline v16h frag_from(const _Float16* c0, const _Float16* c1) {
    v8h lo = *(const v8h*)c0;            // ds_load_b128
    v8h hi = *(const v8h*)c1;            // ds_load_b128
    v16h r;
#pragma unroll
    for (int i = 0; i < 8; ++i) { r[i] = lo[i]; r[i + 8] = hi[i]; }
    return r;
}

extern "C" __global__ __launch_bounds__(NTHREADS)
void boxattn_wmma_kernel(const float* __restrict__ q,
                         const float* __restrict__ k,
                         const float* __restrict__ v,
                         float* __restrict__ out)
{
    extern __shared__ _Float16 lds[];
    _Float16* Qh = lds;
    _Float16* Kh = lds + QH_HALF;
    _Float16* Vt = lds + 2 * QH_HALF;
    _Float16* Wb = lds + 3 * QH_HALF;
    float*    Os = (float*)(lds + OS_OFF);

    const int t    = threadIdx.x;
    const int lane = t & 31;
    const int wv   = t >> 5;                 // wave id == pixel within block
    const int pix0 = blockIdx.x * PIX;
    const int b    = pix0 / HW;              // PIX | HW -> never straddles batch
    const int hw   = pix0 % HW;

    // ---------- stage: coalesced b128 loads, packed f16x2 stores to LDS ------
    // Q/K: row pairs (i, c2),(i, c2+1)  -> h2 pack along ch into Qh/Kh[i][p][c]
    // V  : row pairs (i2, c),(i2+1, c)  -> h2 pack along box into Vt[c][p][j]
#pragma unroll
    for (int it = 0; it < 8; ++it) {
        const int u = it * NTHREADS + t;               // 0..1023
        const int i  = u >> 5, c2 = (u & 31) * 2;
        const size_t gq = ((size_t)(i * NBATCH + b) * CH + c2) * HW + hw;
        v4f q0 = *(const v4f*)(q + gq);
        v4f q1 = *(const v4f*)(q + gq + HW);
        v4f k0 = *(const v4f*)(k + gq);
        v4f k1 = *(const v4f*)(k + gq + HW);
        const int cv = u >> 4, i2 = (u & 15) * 2;
        const size_t gv = ((size_t)(i2 * NBATCH + b) * CH + cv) * HW + hw;
        v4f v0 = *(const v4f*)(v + gv);
        v4f v1 = *(const v4f*)(v + gv + (size_t)NBATCH * CH * HW);
        _Float16* qd = Qh + (i * PIX) * CH + c2;       // +p*128B imm offsets
        _Float16* kd = Kh + (i * PIX) * CH + c2;
        _Float16* vd = Vt + (cv * PIX) * NBOX + i2;    // +p*64B imm offsets
#pragma unroll
        for (int p = 0; p < PIX; ++p) {
            h2 qp; qp.x = (_Float16)q0[p]; qp.y = (_Float16)q1[p];
            h2 kp; kp.x = (_Float16)k0[p]; kp.y = (_Float16)k1[p];
            h2 vp; vp.x = (_Float16)v0[p]; vp.y = (_Float16)v1[p];
            *(h2*)(qd + p * CH)   = qp;                // ds_store_b32
            *(h2*)(kd + p * CH)   = kp;
            *(h2*)(vd + p * NBOX) = vp;
        }
    }
    __syncthreads();

    const int p  = wv;            // this wave's pixel
    const int lm = lane & 15;     // position within 16-lane half
    const int lh = lane >> 4;     // which half

    // ---------- S^T = K Q^T : rows j (VGPRs), cols i (lanes); K=64ch, 2 steps
    v8f acc[2][2] = {};           // [jt][nt]
#pragma unroll
    for (int ks = 0; ks < 2; ++ks) {
        const int c0 = ks * 32;
        v16h afr[2], bfr[2];
#pragma unroll
        for (int jt = 0; jt < 2; ++jt) {              // A rows = box j from Kh
            const _Float16* base = Kh + ((jt * 16 + lm) * PIX + p) * CH + c0 + lh * 8;
            afr[jt] = frag_from(base, base + 16);
        }
#pragma unroll
        for (int nt = 0; nt < 2; ++nt) {              // B cols = box i from Qh
            const _Float16* base = Qh + ((nt * 16 + lm) * PIX + p) * CH + c0 + lh * 8;
            bfr[nt] = frag_from(base, base + 16);
        }
#pragma unroll
        for (int jt = 0; jt < 2; ++jt)
#pragma unroll
            for (int nt = 0; nt < 2; ++nt)
                acc[jt][nt] = __builtin_amdgcn_wmma_f32_16x16x32_f16(
                    false, afr[jt], false, bfr[nt],
                    (short)0, acc[jt][nt], false, false);
    }

    // ---------- softmax over j: register reduction + one xor-16 swizzle ------
    // Lane holds j = jt*16 + lh*8 + r for its column i = nt*16 + lm; the other
    // 16 j's live in the partner lane (lane ^ 16).
#pragma unroll
    for (int nt = 0; nt < 2; ++nt) {
        float vals[16];
#pragma unroll
        for (int jt = 0; jt < 2; ++jt)
#pragma unroll
            for (int r = 0; r < 8; ++r)
                vals[jt * 8 + r] = acc[jt][nt][r] * 0.125f;   // 1/sqrt(64)
        float mx = vals[0];
#pragma unroll
        for (int x = 1; x < 16; ++x) mx = fmaxf(mx, vals[x]);
        mx = fmaxf(mx, SWIZ_XOR(mx, 16));
        float e[16], sm = 0.0f;
#pragma unroll
        for (int x = 0; x < 16; ++x) { e[x] = __expf(vals[x] - mx); sm += e[x]; }
        sm += SWIZ_XOR(sm, 16);
        const float inv = 1.0f / sm;
        const int iW = nt * 16 + lm;
#pragma unroll
        for (int jt = 0; jt < 2; ++jt) {              // 8 consecutive j -> b128
            v8h pk;
#pragma unroll
            for (int r = 0; r < 8; ++r) pk[r] = (_Float16)(e[jt * 8 + r] * inv);
            *(v8h*)(Wb + (p * 32 + iW) * 32 + jt * 16 + lh * 8) = pk;
        }
    }
    // DS ops from one wave are in-order; Wb RAW below is wave-private.

    // ---------- O^T = V^T W^T : rows c (VGPRs), cols i (lanes); K=32 boxes ---
    v16h vafr[4], wbfr[2];
#pragma unroll
    for (int ct = 0; ct < 4; ++ct) {                  // A rows = ch c from Vt
        const _Float16* base = Vt + ((ct * 16 + lm) * PIX + p) * NBOX + lh * 8;
        vafr[ct] = frag_from(base, base + 16);
    }
#pragma unroll
    for (int nt = 0; nt < 2; ++nt) {                  // B cols = box i from Wb
        const _Float16* base = Wb + (p * 32 + nt * 16 + lm) * 32 + lh * 8;
        wbfr[nt] = frag_from(base, base + 16);
    }
    v8f dacc[4][2] = {};
#pragma unroll
    for (int ct = 0; ct < 4; ++ct)
#pragma unroll
        for (int nt = 0; nt < 2; ++nt)
            dacc[ct][nt] = __builtin_amdgcn_wmma_f32_16x16x32_f16(
                false, vafr[ct], false, wbfr[nt],
                (short)0, dacc[ct][nt], false, false);

    // ---------- transpose out through Os, then coalesced b128 stores ---------
    // D element: row c = ct*16 + r + 8*lh, col i = nt*16 + lm.
#pragma unroll
    for (int ct = 0; ct < 4; ++ct)
#pragma unroll
        for (int nt = 0; nt < 2; ++nt)
#pragma unroll
            for (int r = 0; r < 8; ++r) {
                const int c = ct * 16 + r + (lh << 3);
                const int i = nt * 16 + lm;
                Os[(i * CH + c) * PIX + p] = dacc[ct][nt][r];
            }
    __syncthreads();

#pragma unroll
    for (int it = 0; it < 16; ++it) {
        const int r = it * NTHREADS + t;
        const int i = r >> 6, c = r & 63;
        const size_t g = ((size_t)(i * NBATCH + b) * CH + c) * HW + hw;
        *(v4f*)(out + g) = *(const v4f*)(Os + r * 4);  // global_store_b128
    }
}

extern "C" void kernel_launch(void* const* d_in, const int* in_sizes, int n_in,
                              void* d_out, int out_size, void* d_ws, size_t ws_size,
                              hipStream_t stream) {
    const float* q = (const float*)d_in[0];
    const float* k = (const float*)d_in[1];
    const float* v = (const float*)d_in[2];
    float* out = (float*)d_out;

    const int nblocks = (NBATCH * HW) / PIX;                    // 6400
    const size_t shmem = (size_t)OS_OFF * sizeof(_Float16)      // f16 slabs + Wb
                       + (size_t)(NBOX * CH * PIX) * sizeof(float); // Os
    hipLaunchKernelGGL(boxattn_wmma_kernel, dim3(nblocks), dim3(NTHREADS),
                       shmem, stream, q, k, v, out);
}
